// Attention_D_72516227826130
// MI455X (gfx1250) — matrix operand order repaired and checked by reading, not compiled, not run on hardware
//
#include <hip/hip_runtime.h>
#include <hip/hip_bf16.h>

// ---------------------------------------------------------------------------
// Attention with pairwise bias d. B=2, N=256, C=768, H=12, hd=64.
// Bandwidth-bound by the 402 MB one-shot read of d (~17us at 23.3 TB/s);
// d is streamed with non-temporal b128 loads so it never displaces the
// reused tensors in L2. All GEMMs use bf16 v_wmma_f32_16x16x32_bf16.
// ---------------------------------------------------------------------------

typedef __attribute__((ext_vector_type(16))) __bf16 v16bf;
typedef __attribute__((ext_vector_type(8)))  __bf16 v8bf;
typedef __attribute__((ext_vector_type(4)))  __bf16 v4bf;
typedef __attribute__((ext_vector_type(8)))  float  v8f;
typedef __attribute__((ext_vector_type(4)))  float  v4f;

#define BB 2
#define NN 256
#define CC 768
#define HH 12
#define HD 64
#define MM (BB*NN)   // 512 rows (b,n)
#define C3 (3*CC)    // 2304

// ---- WMMA fragment loaders (CDNA5 16x16x32 bf16 layouts, ISA 7.12.2) ------

// A: 16x32 (MxK), row-major source, row stride ld (elements).
// lane L: row m=L&15; half hi=L>>4; K contiguous: hi*16 + e
__device__ __forceinline__ v16bf frag_a(const __bf16* p0, int ld) {
  const int lane = threadIdx.x & 31;
  const int m = lane & 15, hi = lane >> 4;
  const __bf16* p = p0 + m * ld + hi * 16;
  v8bf lo = *(const v8bf*)p;          // K = hi*16 .. hi*16+7
  v8bf h8 = *(const v8bf*)(p + 8);    // K = hi*16+8 .. hi*16+15
  v16bf a;
#pragma unroll
  for (int e = 0; e < 8; ++e) { a[e] = lo[e]; a[8 + e] = h8[e]; }
  return a;
}

// B = W^T  (out = A * W^T): B(k,n) = W[n0+n, k0+k]. p0 -> W[n0,k0], ld = W row stride.
// lane L: col n=L&15; K contiguous: hi*16 + e
__device__ __forceinline__ v16bf frag_b_wt(const __bf16* p0, int ld) {
  const int lane = threadIdx.x & 31;
  const int n = lane & 15, hi = lane >> 4;
  const __bf16* p = p0 + n * ld + hi * 16;
  v8bf lo = *(const v8bf*)p;
  v8bf h8 = *(const v8bf*)(p + 8);
  v16bf b;
#pragma unroll
  for (int e = 0; e < 8; ++e) { b[e] = lo[e]; b[8 + e] = h8[e]; }
  return b;
}

// B taken directly from row-major V: B(k,n) = V[k0+k, n0+n]. p0 -> V[k0,n0].
__device__ __forceinline__ v16bf frag_b_direct(const __bf16* p0, int ld) {
  const int lane = threadIdx.x & 31;
  const int n = lane & 15, hi = lane >> 4;
  v16bf b;
#pragma unroll
  for (int e = 0; e < 16; ++e) b[e] = p0[(hi * 16 + e) * ld + n];
  return b;
}

__device__ __forceinline__ v8f wmma_bf16(v16bf a, v16bf b, v8f c) {
  return __builtin_amdgcn_wmma_f32_16x16x32_bf16(false, a, false, b, (short)0, c,
                                                 false, false);
}

// ---- K0: fp32 -> bf16 convert ---------------------------------------------
__global__ void k_cvt(const float* __restrict__ src, __bf16* __restrict__ dst, int n4) {
  int i = blockIdx.x * blockDim.x + threadIdx.x;
  for (; i < n4; i += gridDim.x * blockDim.x) {
    float4 v = ((const float4*)src)[i];
    v4bf o;
    o[0] = (__bf16)v.x; o[1] = (__bf16)v.y; o[2] = (__bf16)v.z; o[3] = (__bf16)v.w;
    ((v4bf*)dst)[i] = o;
  }
}

// ---- K1: qkv = x @ w_qkv^T ; scatter into q(scaled)/k/v [B,H,N,hd] bf16 ----
__global__ void k_qkv(const __bf16* __restrict__ xb, const __bf16* __restrict__ wqkv,
                      __bf16* __restrict__ qb, __bf16* __restrict__ kb,
                      __bf16* __restrict__ vb) {
  const int wave = threadIdx.x >> 5;
  const int lane = threadIdx.x & 31;
  const int nt = blockIdx.x * 8 + wave;  // 0..143  (2304/16)
  const int mt = blockIdx.y;             // 0..31   (512/16)
  const __bf16* A = xb + (mt * 16) * CC;
  const __bf16* W = wqkv + (nt * 16) * CC;
  v8f acc = {};
#pragma unroll 4
  for (int kk = 0; kk < CC; kk += 32)
    acc = wmma_bf16(frag_a(A + kk, CC), frag_b_wt(W + kk, CC), acc);

  const int hi = lane >> 4, n = lane & 15;
  const int cc = nt * 16 + n;
  const int which = cc / CC;             // 0=q 1=k 2=v (uniform per tile)
  const int c = cc % CC;
  const int h = c / HD, dd = c % HD;
  __bf16* dst = (which == 0) ? qb : (which == 1) ? kb : vb;
  const float scale = (which == 0) ? 0.125f : 1.0f;  // hd^-0.5 = 1/8
#pragma unroll
  for (int vv = 0; vv < 8; ++vv) {
    int r = mt * 16 + vv + 8 * hi;       // global (b,n) row
    int b = r / NN, ns = r % NN;
    dst[((b * HH + h) * NN + ns) * HD + dd] = (__bf16)(acc[vv] * scale);
  }
}

// ---- K2: S = q @ k^T, row softmax -> attn (fp32 + bf16) -------------------
#define SSTR 260
__global__ void k_scores(const __bf16* __restrict__ qb, const __bf16* __restrict__ kb,
                         float* __restrict__ gattn, __bf16* __restrict__ gattnb) {
  __shared__ float sS[16 * SSTR];
  __shared__ float sRed[16 * 16];
  __shared__ float sRow[16];
  const int it = blockIdx.x, h = blockIdx.y, b = blockIdx.z;
  const int bh = b * HH + h;
  const int wave = threadIdx.x >> 5, lane = threadIdx.x & 31;
  const __bf16* A0 = qb + (bh * NN + it * 16) * HD;
#pragma unroll
  for (int t = 0; t < 2; ++t) {
    const int jt = wave + t * 8;         // 16 j-tiles / 8 waves
    const __bf16* W0 = kb + (bh * NN + jt * 16) * HD;
    v8f acc = {};
    acc = wmma_bf16(frag_a(A0, HD),      frag_b_wt(W0, HD),      acc);
    acc = wmma_bf16(frag_a(A0 + 32, HD), frag_b_wt(W0 + 32, HD), acc);
    const int hi = lane >> 4, n = lane & 15;
#pragma unroll
    for (int vv = 0; vv < 8; ++vv)
      sS[(vv + 8 * hi) * SSTR + jt * 16 + n] = acc[vv];
  }
  __syncthreads();

  const int ty = threadIdx.x >> 4;       // row 0..15
  const int tx = threadIdx.x & 15;       // 16-col chunk
  float* row = sS + ty * SSTR;
  float m = -3.0e38f;
#pragma unroll
  for (int e = 0; e < 16; ++e) m = fmaxf(m, row[tx * 16 + e]);
  sRed[ty * 16 + tx] = m;
  __syncthreads();
  if (tx == 0) {
    float mm = sRed[ty * 16];
    for (int e = 1; e < 16; ++e) mm = fmaxf(mm, sRed[ty * 16 + e]);
    sRow[ty] = mm;
  }
  __syncthreads();
  const float rmax = sRow[ty];
  float s = 0.f;
#pragma unroll
  for (int e = 0; e < 16; ++e) {
    float p = __expf(row[tx * 16 + e] - rmax);
    row[tx * 16 + e] = p;
    s += p;
  }
  sRed[ty * 16 + tx] = s;
  __syncthreads();
  if (tx == 0) {
    float ss = 0.f;
    for (int e = 0; e < 16; ++e) ss += sRed[ty * 16 + e];
    sRow[ty] = ss;
  }
  __syncthreads();
  const float inv = 1.0f / sRow[ty];
  float*  go  = gattn  + ((size_t)bh * NN + it * 16 + ty) * NN;
  __bf16* gob = gattnb + ((size_t)bh * NN + it * 16 + ty) * NN;
#pragma unroll
  for (int e = 0; e < 16; ++e) {
    float p = row[tx * 16 + e] * inv;
    go[tx * 16 + e]  = p;
    gob[tx * 16 + e] = (__bf16)p;
  }
}

// ---- K3: out1 = attn @ v  (per b,h) -> fp32 [B,N,C] -----------------------
__global__ void k_av(const __bf16* __restrict__ gattnb, const __bf16* __restrict__ vb,
                     float* __restrict__ gout1) {
  const int it = blockIdx.x, h = blockIdx.y, b = blockIdx.z;
  const int bh = b * HH + h;
  const int wave = threadIdx.x >> 5, lane = threadIdx.x & 31; // 4 waves: col tiles
  const __bf16* A0 = gattnb + ((size_t)bh * NN + it * 16) * NN;
  const __bf16* V0 = vb + (size_t)bh * NN * HD + wave * 16;
  v8f acc = {};
  for (int kk = 0; kk < NN; kk += 32)
    acc = wmma_bf16(frag_a(A0 + kk, NN), frag_b_direct(V0 + kk * HD, HD), acc);
  const int hi = lane >> 4, n = lane & 15;
#pragma unroll
  for (int vv = 0; vv < 8; ++vv) {
    int ns = it * 16 + vv + 8 * hi;
    gout1[((size_t)b * NN + ns) * CC + h * HD + wave * 16 + n] = acc[vv];
  }
}

// ---- K4: stream d (the bandwidth bound):
//   out2[b,i,c] = sum_j attn[b,h(c),i,j] * d[b,i,j,c]
// Non-temporal b128 loads (d is read once; don't pollute L2), j unrolled x4
// for deep outstanding-load queues. Combines with out1, emits bf16 activations.
__global__ void k_dterm(const float* __restrict__ d, const float* __restrict__ gattn,
                        const float* __restrict__ gout1, __bf16* __restrict__ preproj) {
  __shared__ float sAttn[HH * NN];       // 12 KB: attn row i for all heads
  const int bi = blockIdx.x;             // 0..511
  const int b = bi / NN, i = bi % NN;
  for (int idx = threadIdx.x; idx < HH * NN; idx += blockDim.x) {
    int hh = idx >> 8, jj = idx & 255;
    sAttn[idx] = gattn[(((size_t)b * HH + hh) * NN + i) * NN + jj];
  }
  __syncthreads();

  const int c4 = threadIdx.x;            // 0..191 (CC/4 float4 columns)
  const int h = (c4 * 4) / HD;
  const v4f* drow = (const v4f*)(d + ((size_t)(b * NN + i)) * NN * CC) + c4;
  const float* aw = sAttn + h * NN;
  v4f acc = {0.f, 0.f, 0.f, 0.f};
  for (int j = 0; j < NN; j += 4) {
    // 4 independent NT b128 loads -> 2KB outstanding per wave per iteration
    v4f d0 = __builtin_nontemporal_load(drow + (size_t)(j + 0) * (CC / 4));
    v4f d1 = __builtin_nontemporal_load(drow + (size_t)(j + 1) * (CC / 4));
    v4f d2 = __builtin_nontemporal_load(drow + (size_t)(j + 2) * (CC / 4));
    v4f d3 = __builtin_nontemporal_load(drow + (size_t)(j + 3) * (CC / 4));
    const float w0 = aw[j + 0], w1 = aw[j + 1], w2 = aw[j + 2], w3 = aw[j + 3];
    acc += w0 * d0;
    acc += w1 * d1;
    acc += w2 * d2;
    acc += w3 * d3;
  }
  const v4f g = ((const v4f*)(gout1 + (size_t)(b * NN + i) * CC))[c4];
  const v4f r = g + acc;
  v4bf o;
  o[0] = (__bf16)r.x; o[1] = (__bf16)r.y; o[2] = (__bf16)r.z; o[3] = (__bf16)r.w;
  ((v4bf*)(preproj + (size_t)(b * NN + i) * CC))[c4] = o;
}

// ---- K5: d_out = preproj @ w_proj^T + b_proj (fp32 out) -------------------
__global__ void k_proj(const __bf16* __restrict__ preproj, const __bf16* __restrict__ wproj,
                       const float* __restrict__ bias, float* __restrict__ out) {
  const int wave = threadIdx.x >> 5, lane = threadIdx.x & 31;
  const int nt = blockIdx.x * 8 + wave;  // 0..47
  const int mt = blockIdx.y;             // 0..31
  const __bf16* A = preproj + (mt * 16) * CC;
  const __bf16* W = wproj + (nt * 16) * CC;
  v8f acc = {};
#pragma unroll 4
  for (int kk = 0; kk < CC; kk += 32)
    acc = wmma_bf16(frag_a(A + kk, CC), frag_b_wt(W + kk, CC), acc);
  const int hi = lane >> 4, n = lane & 15;
  const float bv = bias[nt * 16 + n];
#pragma unroll
  for (int vv = 0; vv < 8; ++vv) {
    int r = mt * 16 + vv + 8 * hi;
    out[(size_t)r * CC + nt * 16 + n] = acc[vv] + bv;
  }
}

// ---------------------------------------------------------------------------
extern "C" void kernel_launch(void* const* d_in, const int* in_sizes, int n_in,
                              void* d_out, int out_size, void* d_ws, size_t ws_size,
                              hipStream_t stream) {
  const float* x     = (const float*)d_in[0];  // [2,256,768]
  const float* d     = (const float*)d_in[1];  // [2,256,256,768]
  const float* wqkv  = (const float*)d_in[2];  // [2304,768]
  const float* wproj = (const float*)d_in[3];  // [768,768]
  const float* bproj = (const float*)d_in[4];  // [768]
  float* out = (float*)d_out;                  // [2,256,768]

  char* ws = (char*)d_ws;
  size_t off = 0;
  auto alloc = [&](size_t bytes) -> void* {
    void* p = ws + off;
    off = (off + bytes + 255) & ~(size_t)255;
    return p;
  };
  __bf16* xb      = (__bf16*)alloc((size_t)MM * CC * 2);
  __bf16* wqkvb   = (__bf16*)alloc((size_t)C3 * CC * 2);
  __bf16* wprojb  = (__bf16*)alloc((size_t)CC * CC * 2);
  __bf16* qb      = (__bf16*)alloc((size_t)BB * HH * NN * HD * 2);
  __bf16* kb      = (__bf16*)alloc((size_t)BB * HH * NN * HD * 2);
  __bf16* vb      = (__bf16*)alloc((size_t)BB * HH * NN * HD * 2);
  float*  gattn   = (float*)alloc((size_t)BB * HH * NN * NN * 4);
  __bf16* gattnb  = (__bf16*)alloc((size_t)BB * HH * NN * NN * 2);
  float*  gout1   = (float*)alloc((size_t)MM * CC * 4);
  __bf16* preproj = (__bf16*)alloc((size_t)MM * CC * 2);

  k_cvt<<<256, 256, 0, stream>>>(x, xb, MM * CC / 4);
  k_cvt<<<512, 256, 0, stream>>>(wqkv, wqkvb, C3 * CC / 4);
  k_cvt<<<256, 256, 0, stream>>>(wproj, wprojb, CC * CC / 4);
  k_qkv  <<<dim3(18, 32),      256, 0, stream>>>(xb, wqkvb, qb, kb, vb);
  k_scores<<<dim3(16, HH, BB), 256, 0, stream>>>(qb, kb, gattn, gattnb);
  k_av   <<<dim3(16, HH, BB), 128, 0, stream>>>(gattnb, vb, gout1);
  k_dterm<<<512, 192, 0, stream>>>(d, gattn, gout1, preproj);
  k_proj <<<dim3(6, 32),      256, 0, stream>>>(preproj, wprojb, bproj, out);
}
